// DeepEZGATFiLMHemiSSDB_82033875354143
// MI455X (gfx1250) — matrix operand order repaired and checked by reading, compiled, not run on hardware
//
#include <hip/hip_runtime.h>

#define NNODES 4096
#define NEG 0.1f
#define MASKF -9.0e15f

typedef __attribute__((ext_vector_type(16))) __bf16 v16bf;
typedef __attribute__((ext_vector_type(2)))  __bf16 v2bf;
typedef __attribute__((ext_vector_type(8)))  float  v8f;
typedef unsigned short u16;
typedef unsigned int   u32;

union BF16x16 { v16bf v; v2bf p[8]; u32 w[8]; u16 u[16]; uint4 q[2]; };
union F16x    { float4 q[4]; float f[16]; };

__device__ __forceinline__ float leakyf(float x) { return x > 0.f ? x : NEG * x; }
__device__ __forceinline__ u16 f2bf(float x) {
  u32 u = __float_as_uint(x);
  u32 r = u + 0x7FFFu + ((u >> 16) & 1u);
  return (u16)(r >> 16);
}
// Pack 16 floats -> 16 bf16. Preferred: packed HW convert; otherwise
// round(+0x7FFF) and pack two high halves per v_perm_b32 (1.5 VALU/element).
__device__ __forceinline__ void pack16(BF16x16& d, const float* f) {
#if __has_builtin(__builtin_amdgcn_cvt_pk_bf16_f32)
#pragma unroll
  for (int i = 0; i < 8; ++i)
    d.p[i] = __builtin_amdgcn_cvt_pk_bf16_f32(f[2 * i], f[2 * i + 1]);
#else
#pragma unroll
  for (int i = 0; i < 8; ++i) {
    u32 lo = __float_as_uint(f[2 * i]) + 0x7FFFu;
    u32 hi = __float_as_uint(f[2 * i + 1]) + 0x7FFFu;
    // dst[15:0] = lo[31:16], dst[31:16] = hi[31:16]
    d.w[i] = __builtin_amdgcn_perm(hi, lo, 0x07060302u);
  }
#endif
}

// ---------------------------------------------------------------------------
// Async global->LDS staging (CDNA5 ASYNCcnt path).
// Copies CNT b128 segments per thread from a contiguous global block into LDS.
// ---------------------------------------------------------------------------
template <int CNT>
__device__ __forceinline__ void stage_chunk(const u16* __restrict__ g, u16* l) {
  const int tid = threadIdx.x;
#pragma unroll
  for (int i = 0; i < CNT; ++i) {
    int s = tid + i * 256;
    asm volatile("global_load_async_to_lds_b128 %0, %1, off"
                 :
                 : "v"((__attribute__((address_space(3))) u16*)(l + s * 8)),
                   "v"(g + s * 8)
                 : "memory");
  }
}
template <int N>
__device__ __forceinline__ void wait_async() {
  asm volatile("s_wait_asynccnt %0" : : "i"(N) : "memory");
}

// ---------------------------------------------------------------------------
// Packing kernels
// ---------------------------------------------------------------------------
// adj (int32 0/1) -> bitmask [4096][128] words, bit j&31 of word j>>5
__global__ void pack_adj_kernel(const int* __restrict__ adj, u32* __restrict__ bits) {
  int idx = blockIdx.x * blockDim.x + threadIdx.x;  // 4096*128
  if (idx >= NNODES * 128) return;
  const int4* p = (const int4*)(adj + (size_t)idx * 32);
  u32 m = 0;
#pragma unroll
  for (int q = 0; q < 8; ++q) {
    int4 v = p[q];
    m |= (v.x > 0 ? 1u : 0u) << (q * 4 + 0);
    m |= (v.y > 0 ? 1u : 0u) << (q * 4 + 1);
    m |= (v.z > 0 ? 1u : 0u) << (q * 4 + 2);
    m |= (v.w > 0 ? 1u : 0u) << (q * 4 + 3);
  }
  bits[idx] = m;
}
// W1 (4,4096,30) -> bf16 [4096][128], col c = h*30+o, pad 120..127 = 0
__global__ void pack_w1_kernel(const float* __restrict__ W1, u16* __restrict__ W1bf) {
  int k = blockIdx.x * blockDim.x + threadIdx.x;
  if (k >= 4096) return;
  for (int c = 0; c < 128; ++c) {
    u16 v = 0;
    if (c < 120) {
      int h = c / 30, o = c % 30;
      v = f2bf(W1[(size_t)h * 4096 * 30 + (size_t)k * 30 + o]);
    }
    W1bf[(size_t)k * 128 + c] = v;
  }
}
// W2 (1,120,50) -> bf16 [128][64] zero padded
__global__ void pack_w2_kernel(const float* __restrict__ W2, u16* __restrict__ W2bf) {
  int k = threadIdx.x;  // 128
  for (int c = 0; c < 64; ++c) {
    u16 v = 0;
    if (k < 120 && c < 50) v = f2bf(W2[(size_t)k * 50 + c]);
    W2bf[(size_t)k * 64 + c] = v;
  }
}
// zero pad columns 120..127 of h1f [4096][128]
__global__ void zero_h1pad_kernel(float* __restrict__ h1f) {
  int n = blockIdx.x * blockDim.x + threadIdx.x;
  if (n >= NNODES) return;
  for (int c = 120; c < 128; ++c) h1f[(size_t)n * 128 + c] = 0.f;
}

// ---------------------------------------------------------------------------
// bf16 WMMA GEMM:  C[M][ldc] = A(f32)[M][lda] x B(bf16)[K][TN*16]
// block = 256 thr (8 waves); B panel staged in LDS via async DMA, double-buffered
// in 128-row K-chunks. Wave tile 16 x (TN*16).
// ---------------------------------------------------------------------------
template <int TN>
__global__ __launch_bounds__(256) void gemm_bf16_kernel(
    const float* __restrict__ A, const u16* __restrict__ B, float* __restrict__ C,
    int K, int lda, int ldc) {
  constexpr int LDB = TN * 16;
  constexpr int CHUNK = 128 * LDB;  // elements per K-chunk
  __shared__ __align__(16) u16 bsm[2 * CHUNK];

  const int lane = threadIdx.x & 31;
  const int wave = threadIdx.x >> 5;
  const int m0 = blockIdx.x * 128 + wave * 16;
  const int hi = lane >> 4;  // 0/1 half of wave
  const int kb = hi * 8;     // K sub-base per A layout
  const int mr = lane & 15;
  const int nchunk = K >> 7;

  v8f acc[TN];
#pragma unroll
  for (int t = 0; t < TN; ++t) acc[t] = {};

  const float* arow = A + (size_t)(m0 + mr) * lda;

  stage_chunk<TN>(B, bsm);  // prologue: chunk 0 -> buf 0
  for (int c = 0; c < nchunk; ++c) {
    if (c + 1 < nchunk) {
      stage_chunk<TN>(B + (size_t)(c + 1) * CHUNK, bsm + ((c + 1) & 1) * CHUNK);
      wait_async<TN>();  // chunk c complete (in-order), c+1 may be in flight
    } else {
      wait_async<0>();
    }
    __syncthreads();
    const u16* bbuf = bsm + (c & 1) * CHUNK;
    const int kc = c << 7;

#pragma unroll
    for (int kk = 0; kk < 128; kk += 32) {
      __builtin_prefetch(arow + kc + kk + 256, 0, 1);  // global_prefetch_b8
      // A fragment: 16x32 bf16, ISA layout (lane pair per row, split K runs)
      F16x af;
      af.q[0] = *(const float4*)(arow + kc + kk + kb);
      af.q[1] = *(const float4*)(arow + kc + kk + kb + 4);
      af.q[2] = *(const float4*)(arow + kc + kk + 16 + kb);
      af.q[3] = *(const float4*)(arow + kc + kk + 16 + kb + 4);
      BF16x16 abf;
      pack16(abf, af.f);

      // B fragments from LDS: lane L holds column N=L%16, K in table order (8h+s, 16+8h+s)
      BF16x16 bb[TN];
#pragma unroll
      for (int t = 0; t < TN; ++t) {
        const u16* bp = bbuf + (kk + kb) * LDB + t * 16 + (lane & 15);
#pragma unroll
        for (int s = 0; s < 8; ++s) {
          bb[t].u[s] = bp[s * LDB];
          bb[t].u[8 + s] = bp[(16 + s) * LDB];
        }
      }
#pragma unroll
      for (int t = 0; t < TN; ++t)
        acc[t] = __builtin_amdgcn_wmma_f32_16x16x32_bf16(
            false, abf.v, false, bb[t].v, (short)0, acc[t], false, false);
    }
    __syncthreads();  // all waves done reading buf (c&1) before it is re-filled
  }
#pragma unroll
  for (int t = 0; t < TN; ++t)
#pragma unroll
    for (int v = 0; v < 8; ++v) {
      int r = v + hi * 8;
      C[(size_t)(m0 + r) * ldc + t * 16 + (lane & 15)] = acc[t][v];
    }
}

// ---------------------------------------------------------------------------
// Attention-coefficient prep: f1/f2 dots + bf16 copy of wh (padded)
// ---------------------------------------------------------------------------
__global__ void prep_wh1_kernel(const float* __restrict__ whpad,
                                const float* __restrict__ a1s, const float* __restrict__ a1d,
                                u16* __restrict__ whbf, float* __restrict__ f1,
                                float* __restrict__ f2) {
  int t = blockIdx.x * blockDim.x + threadIdx.x;
  if (t >= 4 * NNODES) return;
  int h = t >> 12, n = t & 4095;
  float s1 = 0.f, s2 = 0.f;
  u16* dst = whbf + ((size_t)h * NNODES + n) * 32;
  const float* src = whpad + (size_t)n * 128 + h * 30;
  for (int o = 0; o < 30; ++o) {
    float w = src[o];
    s1 += w * a1s[h * 30 + o];
    s2 += w * a1d[h * 30 + o];
    dst[o] = f2bf(w);
  }
  dst[30] = 0; dst[31] = 0;
  f1[h * NNODES + n] = s1;
  f2[h * NNODES + n] = s2;
}
__global__ void prep_wh2_kernel(const float* __restrict__ whpad,
                                const float* __restrict__ a2s, const float* __restrict__ a2d,
                                u16* __restrict__ whbf, float* __restrict__ f1,
                                float* __restrict__ f2) {
  int n = blockIdx.x * blockDim.x + threadIdx.x;
  if (n >= NNODES) return;
  float s1 = 0.f, s2 = 0.f;
  u16* dst = whbf + (size_t)n * 64;
  const float* src = whpad + (size_t)n * 64;
  for (int o = 0; o < 50; ++o) {
    float w = src[o];
    s1 += w * a2s[o];
    s2 += w * a2d[o];
    dst[o] = f2bf(w);
  }
  for (int o = 50; o < 64; ++o) dst[o] = 0;
  f1[n] = s1;
  f2[n] = s2;
}

// ---------------------------------------------------------------------------
// Fused masked-softmax attention aggregation (flash-style online softmax).
// One wave owns (head, 16-row strip). Mask comes from the packed bitmask
// (1 b32 word per lane per 32-wide tile). WH panel is staged in LDS via async
// DMA, double-buffered in 128-row chunks shared by all 8 waves of the block.
// ---------------------------------------------------------------------------
template <int TN>
__global__ __launch_bounds__(256) void attn_kernel(
    const u16* __restrict__ whbf, const float* __restrict__ f1, const float* __restrict__ f2,
    const u32* __restrict__ adjbits, const float* __restrict__ bias, float* __restrict__ out,
    int ldout, int olimit) {
  constexpr int LDB = TN * 16;
  constexpr int CHUNK = 128 * LDB;
  __shared__ __align__(16) u16 bsm[2 * CHUNK];

  const int lane = threadIdx.x & 31;
  const int wv = threadIdx.x >> 5;
  const int gw = blockIdx.x * 8 + wv;
  const int head = gw >> 8;  // 256 row-strips per head; 8 waves of a block share head
  const int i0 = (gw & 255) * 16;
  const int r = lane & 15;
  const int hi = lane >> 4;
  const int kb = hi * 8;
  const int ocol0 = head * olimit;

  const float f1i = f1[head * NNODES + i0 + r];
  const float* f2h = f2 + head * NNODES;
  const u32* mrow = adjbits + (size_t)(i0 + r) * 128;
  const u16* whh = whbf + (size_t)head * NNODES * LDB;

  v8f acc[TN];
#pragma unroll
  for (int t = 0; t < TN; ++t) acc[t] = {};
  float m = -__builtin_inff();
  float l = 0.f;

  stage_chunk<TN>(whh, bsm);  // chunk 0
  for (int c = 0; c < 32; ++c) {  // 4096 / 128 chunks over j
    if (c + 1 < 32) {
      stage_chunk<TN>(whh + (size_t)(c + 1) * CHUNK, bsm + ((c + 1) & 1) * CHUNK);
      wait_async<TN>();
    } else {
      wait_async<0>();
    }
    __syncthreads();
    const u16* bbuf = bsm + (c & 1) * CHUNK;

#pragma unroll
    for (int jj = 0; jj < 128; jj += 32) {
      const int j0 = (c << 7) + jj;
      F16x fj;
      fj.q[0] = *(const float4*)(f2h + j0 + kb);
      fj.q[1] = *(const float4*)(f2h + j0 + kb + 4);
      fj.q[2] = *(const float4*)(f2h + j0 + 16 + kb);
      fj.q[3] = *(const float4*)(f2h + j0 + 16 + kb + 4);
      const u32 mw = mrow[j0 >> 5];  // mask bits for j0..j0+31 of this row

      float e[16];
      float mloc = MASKF;
#pragma unroll
      for (int i = 0; i < 16; ++i) {
        int bitpos = (i < 8) ? (kb + i) : (16 + kb + (i - 8));
        float ee = ((mw >> bitpos) & 1u) ? leakyf(f1i + fj.f[i]) : MASKF;
        e[i] = ee;
        mloc = fmaxf(mloc, ee);
      }
      mloc = fmaxf(mloc, __shfl_xor(mloc, 16, 32));  // row lives on lane pair {r, r+16}
      float newm = fmaxf(m, mloc);
      float sc = __expf(m - newm);

      float pv[16];
      float s = 0.f;
#pragma unroll
      for (int i = 0; i < 16; ++i) {
        float p = __expf(e[i] - newm);
        s += p;
        pv[i] = p;
      }
      BF16x16 pbf;
      pack16(pbf, pv);
      s += __shfl_xor(s, 16, 32);
      l = l * sc + s;
      m = newm;

      // rescale accumulators: C-layout row = v + hi*8 ; per-row scale on lane==row
      float scv[8];
#pragma unroll
      for (int v = 0; v < 8; ++v) scv[v] = __shfl(sc, v + hi * 8, 32);
#pragma unroll
      for (int t = 0; t < TN; ++t)
#pragma unroll
        for (int v = 0; v < 8; ++v) acc[t][v] *= scv[v];

      BF16x16 bb[TN];
#pragma unroll
      for (int t = 0; t < TN; ++t) {
        const u16* bp = bbuf + (jj + kb) * LDB + t * 16 + (lane & 15);
#pragma unroll
        for (int s = 0; s < 8; ++s) {
          bb[t].u[s] = bp[s * LDB];
          bb[t].u[8 + s] = bp[(16 + s) * LDB];
        }
      }
#pragma unroll
      for (int t = 0; t < TN; ++t)
        acc[t] = __builtin_amdgcn_wmma_f32_16x16x32_bf16(
            false, pbf.v, false, bb[t].v, (short)0, acc[t], false, false);
    }
    __syncthreads();
  }

  float lr[8];
#pragma unroll
  for (int v = 0; v < 8; ++v) lr[v] = __shfl(l, v + hi * 8, 32);
#pragma unroll
  for (int t = 0; t < TN; ++t)
#pragma unroll
    for (int v = 0; v < 8; ++v) {
      int o = t * 16 + (lane & 15);
      if (o < olimit) {
        int row = i0 + v + hi * 8;
        float val = leakyf(acc[t][v] / lr[v] + bias[ocol0 + o]);
        out[(size_t)row * ldout + ocol0 + o] = val;
      }
    }
}

// ---------------------------------------------------------------------------
// FiLM + gating + classifier logits (per node)
// ---------------------------------------------------------------------------
__global__ void film_kernel(const float* __restrict__ l_loc, const float* __restrict__ h2,
                            const float* __restrict__ Wl1, const float* __restrict__ bl1,
                            const float* __restrict__ Wl2, const float* __restrict__ bl2,
                            const float* __restrict__ Wg, const float* __restrict__ bg,
                            const float* __restrict__ Wb, const float* __restrict__ bb,
                            const float* __restrict__ Wgate, const float* __restrict__ bgate,
                            const float* __restrict__ Wc, float* __restrict__ logits) {
  int n = blockIdx.x * blockDim.x + threadIdx.x;
  if (n >= NNODES) return;
  float ll[16];
  for (int k = 0; k < 16; ++k) ll[k] = l_loc[(size_t)n * 16 + k];
  float z1[32];
  for (int a = 0; a < 32; ++a) {
    float s = bl1[a];
    for (int k = 0; k < 16; ++k) s += Wl1[a * 16 + k] * ll[k];
    z1[a] = leakyf(s);
  }
  float z2[32];
  for (int a = 0; a < 32; ++a) {
    float s = bl2[a];
    for (int k = 0; k < 32; ++k) s += Wl2[a * 32 + k] * z1[k];
    z2[a] = leakyf(s);
  }
  float gs = bgate[0];
  for (int z = 0; z < 32; ++z) gs += Wgate[z] * z2[z];
  float gate = 1.f / (1.f + __expf(-gs));
  float cov = fminf(fmaxf(ll[15], 0.f), 1.f);
  float g = gate * cov;
  float lo0 = 0.f, lo1 = 0.f;
  for (int o = 0; o < 50; ++o) {
    float gam = bg[o], bet = bb[o];
    for (int z = 0; z < 32; ++z) {
      gam += Wg[o * 32 + z] * z2[z];
      bet += Wb[o * 32 + z] * z2[z];
    }
    float hv = h2[(size_t)n * 64 + o];
    float hf = hv + g * (gam * hv + bet);
    lo0 += hf * Wc[o];
    lo1 += hf * Wc[50 + o];
  }
  logits[(size_t)n * 2 + 0] = leakyf(lo0);
  logits[(size_t)n * 2 + 1] = leakyf(lo1);
}

// ---------------------------------------------------------------------------
// Hemispheric SSDB head: b_arr[region*2+c] = leaky(h @ Ws2 + bs2)
// ---------------------------------------------------------------------------
__global__ void ssdb_kernel(const float* __restrict__ logits, const float* __restrict__ Ws1,
                            const float* __restrict__ bs1, const float* __restrict__ Ws2,
                            const float* __restrict__ bs2, float* __restrict__ b_arr) {
  __shared__ float hsh[2][2][60];
  int t = threadIdx.x;
  if (t < 240) {
    int region = t / 120, rem = t % 120, c = rem / 60, k = rem % 60;
    const float* base = logits + (size_t)region * 2048 * 2 + c;
    float s = bs1[k];
    for (int j = 0; j < 2048; ++j) s += base[(size_t)j * 2] * Ws1[(size_t)k * 2048 + j];
    hsh[region][c][k] = leakyf(s);
  }
  __syncthreads();
  if (t < 4) {
    int region = t >> 1, c = t & 1;
    float s = bs2[0];
    for (int k = 0; k < 60; ++k) s += hsh[region][c][k] * Ws2[k];
    b_arr[region * 2 + c] = leakyf(s);
  }
}

__global__ void finalize_kernel(const float* __restrict__ logits, const float* __restrict__ b_arr,
                                float* __restrict__ out) {
  int idx = blockIdx.x * blockDim.x + threadIdx.x;
  if (idx < 2 * NNODES) {
    int n = idx >> 1, c = idx & 1;
    out[idx] = logits[idx] + b_arr[((n >= 2048) ? 2 : 0) + c];
  }
  if (idx < 2) out[2 * NNODES + idx] = 0.5f * (b_arr[idx] + b_arr[2 + idx]);
}

// ---------------------------------------------------------------------------
extern "C" void kernel_launch(void* const* d_in, const int* in_sizes, int n_in,
                              void* d_out, int out_size, void* d_ws, size_t ws_size,
                              hipStream_t stream) {
  const float* x_fc = (const float*)d_in[0];
  const int*   adj  = (const int*)d_in[1];
  const float* lloc = (const float*)d_in[2];
  const float* W1   = (const float*)d_in[3];
  const float* a1s  = (const float*)d_in[4];
  const float* a1d  = (const float*)d_in[5];
  const float* b1   = (const float*)d_in[6];
  const float* W2   = (const float*)d_in[7];
  const float* a2s  = (const float*)d_in[8];
  const float* a2d  = (const float*)d_in[9];
  const float* b2   = (const float*)d_in[10];
  const float* Wc   = (const float*)d_in[11];
  const float* Wl1  = (const float*)d_in[12];
  const float* bl1  = (const float*)d_in[13];
  const float* Wl2  = (const float*)d_in[14];
  const float* bl2  = (const float*)d_in[15];
  const float* Wg   = (const float*)d_in[16];
  const float* bg   = (const float*)d_in[17];
  const float* Wb   = (const float*)d_in[18];
  const float* bb   = (const float*)d_in[19];
  const float* Wgate = (const float*)d_in[20];
  const float* bgate = (const float*)d_in[21];
  const float* Ws1  = (const float*)d_in[22];
  const float* bs1  = (const float*)d_in[23];
  const float* Ws2  = (const float*)d_in[24];
  const float* bs2  = (const float*)d_in[25];
  float* out = (float*)d_out;

  char* w = (char*)d_ws;
  auto alloc = [&](size_t bytes) -> void* {
    void* p = (void*)w;
    w += (bytes + 255) & ~(size_t)255;
    return p;
  };
  float* whpad   = (float*)alloc((size_t)4096 * 128 * 4);   // GEMM1 out
  u16*   whbf    = (u16*)  alloc((size_t)4 * 4096 * 32 * 2);
  float* f1      = (float*)alloc((size_t)4 * 4096 * 4);
  float* f2      = (float*)alloc((size_t)4 * 4096 * 4);
  u16*   W1bf    = (u16*)  alloc((size_t)4096 * 128 * 2);
  float* h1f     = (float*)alloc((size_t)4096 * 128 * 4);   // layer1 out (padded)
  u16*   W2bf    = (u16*)  alloc((size_t)128 * 64 * 2);
  float* wh2pad  = (float*)alloc((size_t)4096 * 64 * 4);    // GEMM2 out
  u16*   wh2bf   = (u16*)  alloc((size_t)4096 * 64 * 2);
  float* f1b     = (float*)alloc((size_t)4096 * 4);
  float* f2b     = (float*)alloc((size_t)4096 * 4);
  float* h2      = (float*)alloc((size_t)4096 * 64 * 4);
  float* logits  = (float*)alloc((size_t)4096 * 2 * 4);
  float* b_arr   = (float*)alloc(16);
  u32*   adjbits = (u32*)  alloc((size_t)4096 * 128 * 4);   // packed mask (2 MB)

  pack_adj_kernel<<<2048, 256, 0, stream>>>(adj, adjbits);
  pack_w1_kernel<<<16, 256, 0, stream>>>(W1, W1bf);
  pack_w2_kernel<<<1, 128, 0, stream>>>(W2, W2bf);
  zero_h1pad_kernel<<<16, 256, 0, stream>>>(h1f);

  // GEMM1: 4096x4096x128(pad of 120), B staged in LDS via async DMA
  gemm_bf16_kernel<8><<<32, 256, 0, stream>>>(x_fc, W1bf, whpad, 4096, 4096, 128);
  prep_wh1_kernel<<<64, 256, 0, stream>>>(whpad, a1s, a1d, whbf, f1, f2);

  // Layer-1 fused attention: 4 heads * 256 strips / 8 waves = 128 blocks
  attn_kernel<2><<<128, 256, 0, stream>>>(whbf, f1, f2, adjbits, b1, h1f, 128, 30);

  // GEMM2: 4096x128x64(pad of 50)
  gemm_bf16_kernel<4><<<32, 256, 0, stream>>>(h1f, W2bf, wh2pad, 128, 128, 64);
  prep_wh2_kernel<<<16, 256, 0, stream>>>(wh2pad, a2s, a2d, wh2bf, f1b, f2b);

  // Layer-2 fused attention: 1 head
  attn_kernel<4><<<32, 256, 0, stream>>>(wh2bf, f1b, f2b, adjbits, b2, h2, 64, 50);

  film_kernel<<<16, 256, 0, stream>>>(lloc, h2, Wl1, bl1, Wl2, bl2, Wg, bg, Wb, bb,
                                      Wgate, bgate, Wc, logits);
  ssdb_kernel<<<1, 256, 0, stream>>>(logits, Ws1, bs1, Ws2, bs2, b_arr);
  finalize_kernel<<<32, 256, 0, stream>>>(logits, b_arr, out);
}
